// VQVAE_37529424233099
// MI455X (gfx1250) — matrix operand order repaired and checked by reading, not compiled, not run on hardware
//
#include <hip/hip_runtime.h>
#include <cstdint>

typedef __attribute__((ext_vector_type(16))) _Float16 v16h;
typedef __attribute__((ext_vector_type(8)))  float    v8f;
typedef __attribute__((ext_vector_type(2)))  float    v2f;
typedef __attribute__((ext_vector_type(4)))  unsigned v4u;
typedef __attribute__((ext_vector_type(8)))  int      v8i;
typedef __attribute__((ext_vector_type(4)))  int      v4i;

// ---------------------------------------------------------------- conv1
// imgs (32,3,256,256) -> h (32,16,128,128), k=4 s=2 p=1, +bias, ReLU. K=48: direct.
__global__ void k_conv1(const float* __restrict__ img, const float* __restrict__ w1,
                        const float* __restrict__ b1, float* __restrict__ out) {
  __shared__ float ws[16 * 48];
  for (int i = threadIdx.x; i < 768; i += blockDim.x) ws[i] = w1[i];
  __syncthreads();
  int idx = blockIdx.x * blockDim.x + threadIdx.x;   // 32*16*128*128
  int ox = idx & 127; int t = idx >> 7;
  int oy = t & 127;   t >>= 7;
  int oc = t & 15;    int b = t >> 4;
  float acc = b1[oc];
  const float* ib = img + (size_t)b * 3 * 256 * 256;
  const float* wp = ws + oc * 48;
  #pragma unroll
  for (int ic = 0; ic < 3; ++ic)
    #pragma unroll
    for (int ky = 0; ky < 4; ++ky) {
      int iy = 2 * oy - 1 + ky;
      if ((unsigned)iy >= 256u) continue;
      #pragma unroll
      for (int kx = 0; kx < 4; ++kx) {
        int ix = 2 * ox - 1 + kx;
        if ((unsigned)ix >= 256u) continue;
        acc = fmaf(ib[(ic * 256 + iy) * 256 + ix], wp[ic * 16 + ky * 4 + kx], acc);
      }
    }
  out[idx] = fmaxf(acc, 0.0f);
}

// ---------------------------------------------------------------- conv2 (WMMA f16)
// h (32,16,128,128) -> z_e (32,32,64,64). Implicit GEMM: M=131072 px, N=32, K=256.
// Wave = 16px (one row segment) x 16ch. Patch staged in LDS, then ds-only fragments.
__global__ void k_conv2_wmma(const float* __restrict__ h, const float* __restrict__ w2,
                             const float* __restrict__ b2, float* __restrict__ ze) {
  __shared__ _Float16 wsh[32 * 256];                 // 16KB: wsh[oc*256 + k]
  __shared__ _Float16 atile[8][16][4][36];           // 36KB: per-wave im2col patch
  for (int i = threadIdx.x; i < 32 * 256; i += blockDim.x)
    wsh[i] = (_Float16)w2[i];                        // (oc,ic,ky,kx) == oc*256+k
  const int lane  = threadIdx.x & 31;
  const int wv    = threadIdx.x >> 5;
  const int tile  = blockIdx.x * 8 + wv;             // 16384 tiles
  const int ntile = tile & 1;
  const int mtile = tile >> 1;
  const int p0 = mtile * 16;                         // 16 px, same b / row (64%16==0)
  const int b  = p0 >> 12;
  const int oy = (p0 >> 6) & 63;
  const int x0 = p0 & 63;
  const float* hb = h + (size_t)b * 16 * 128 * 128;
  const int iy0 = 2 * oy - 1, ix0 = 2 * x0 - 1;
  __builtin_prefetch(&hb[(iy0 > 0 ? iy0 : 0) * 128], 0, 1);
  // fill patch: 16ch x 4 rows x 36 cols (cols 0..33 used), coalesced, no divergence
  for (int i = lane; i < 16 * 4 * 36; i += 32) {
    int col = i % 36; int t = i / 36; int r = t & 3; int ic = t >> 2;
    int iy = iy0 + r, ix = ix0 + col;
    float v = 0.f;
    if ((unsigned)iy < 128u && (unsigned)ix < 128u)
      v = hb[(ic * 128 + iy) * 128 + ix];
    atile[wv][ic][r][col] = (_Float16)v;
  }
  __syncthreads();
  const int lm = lane & 15, hi = lane >> 4;
  const int oc = ntile * 16 + lm;
  v8f c = {};
  for (int kb = 0; kb < 256; kb += 32) {
    v16h a;
    #pragma unroll
    for (int j = 0; j < 16; ++j) {
      int k = kb + hi * 16 + j;                      // A frag K order (contiguous, matches B)
      int ic = k >> 4, tt = k & 15, ky = tt >> 2, kx = tt & 3;
      a[j] = atile[wv][ic][ky][2 * lm + kx];
    }
    v16h bf = *(const v16h*)&wsh[oc * 256 + kb + hi * 16];  // 16 contiguous halves
    c = __builtin_amdgcn_wmma_f32_16x16x32_f16(false, a, false, bf, (short)0, c, false, false);
  }
  const float bias = b2[oc];
  const int rem0 = p0 & 4095;
  #pragma unroll
  for (int j = 0; j < 8; ++j) {
    int m = j + hi * 8;
    ze[((b * 32 + oc) << 12) + rem0 + m] = fmaxf(c[j] + bias, 0.0f);
  }
}

// ---------------------------------------------------------------- VQ (WMMA f32 + TDM)
// dist = ||c||^2 - 2 z.c. Codebook (512x32 f32 = 64KB) DMA'd to LDS by the TDM.
__global__ void k_vq(const float* __restrict__ ze, const float* __restrict__ cb,
                     float* __restrict__ enc, float* __restrict__ loss) {
  __shared__ float csh[512 * 32];                    // 64KB codebook
  __shared__ float zsh[16][32];
  __shared__ float cn[512];
  __shared__ float dist[16][512];
  __shared__ float bd[16][16];
  __shared__ int   bk[16][16];
  __shared__ float lsum[256];

  const int tid = threadIdx.x;
  const int mtile = blockIdx.x;                      // 8192 blocks x 16 pixels
  for (int i = tid; i < 16 * 32; i += 256) {
    int m = i >> 5, d = i & 31;
    int p = mtile * 16 + m;
    int b = p >> 12, rem = p & 4095;
    zsh[m][d] = ze[((b * 32 + d) << 12) + rem];
  }
#if __has_builtin(__builtin_amdgcn_tensor_load_to_lds)
  if (tid == 0) {
    unsigned lds = (unsigned)(uintptr_t)(void*)&csh[0];       // low 32b = LDS offset
    unsigned long long ga = (unsigned long long)(uintptr_t)cb;
    // D# group0: count=1 | lds_addr | global_addr[56:0] | type=2
    v4u g0 = { 1u, lds, (unsigned)(ga & 0xffffffffu),
               (unsigned)((ga >> 32) & 0x1ffffffu) | (2u << 30) };
    // D# group1: data_size=4B; tensor_dim0=32, tensor_dim1=512;
    //            tile_dim0=32, tile_dim1=512; tensor_dim0_stride=32
    v8i g1 = { (int)(2u << 16),      // [17:16] data_size=2 (4 bytes)
               (int)(32u << 16),     // [63:48] tensor_dim0 lo16 = 32
               (int)(512u << 16),    // [95:80] tensor_dim1 lo16 = 512
               (int)(32u << 16),     // [127:112] tile_dim0 = 32
               (int)512,             // [143:128] tile_dim1 = 512
               (int)32,              // [191:160] tensor_dim0_stride lo = 32
               0, 0 };
    v4i g2 = { 0, 0, 0, 0 };
    v4i g3 = { 0, 0, 0, 0 };
    v8i g4 = { 0, 0, 0, 0, 0, 0, 0, 0 };
    __builtin_amdgcn_tensor_load_to_lds(g0, g1, g2, g3, g4, 0);
    __builtin_amdgcn_s_wait_tensorcnt(0);
  }
#else
  for (int i = tid; i < 512 * 32; i += 256) csh[i] = cb[i];
#endif
  __syncthreads();
  for (int k = tid; k < 512; k += 256) {             // codebook norms from LDS
    float s = 0.f;
    const float* cp = &csh[k * 32];
    #pragma unroll
    for (int d = 0; d < 32; ++d) s = fmaf(cp[d], cp[d], s);
    cn[k] = s;
  }
  __syncthreads();

  const int lane = tid & 31;
  const int wv   = tid >> 5;
  const int hi   = lane >> 4;
  const int lm   = lane & 15;
  for (int nt = 0; nt < 4; ++nt) {
    const int code0 = wv * 64 + nt * 16;
    v8f c = {};
    #pragma unroll
    for (int k0 = 0; k0 < 32; k0 += 4) {
      int ka = k0 + hi * 2;                          // f32 A frag: V0=K0/K2, V1=K1/K3
      v2f a = *(const v2f*)&zsh[lm][ka];             // 8B-aligned ds_load_b64
      v2f b = *(const v2f*)&csh[(code0 + lm) * 32 + ka];
      c = __builtin_amdgcn_wmma_f32_16x16x4_f32(false, a, false, b, (short)0, c, false, false);
    }
    int code = code0 + lm;
    float cno = cn[code];
    #pragma unroll
    for (int j = 0; j < 8; ++j)
      dist[j + hi * 8][code] = cno - 2.0f * c[j];
  }
  __syncthreads();
  { // argmin stage 1: 16 threads per pixel
    int m = tid >> 4, st = tid & 15;
    float best = 3.4e38f; int bi = 0;
    for (int k = st; k < 512; k += 16) {
      float dv = dist[m][k];
      if (dv < best) { best = dv; bi = k; }
    }
    bd[m][st] = best; bk[m][st] = bi;
  }
  __syncthreads();
  if ((tid & 15) == 0) {                             // argmin stage 2
    int m = tid >> 4;
    float best = bd[m][0]; int bi = bk[m][0];
    #pragma unroll
    for (int i = 1; i < 16; ++i)
      if (bd[m][i] < best) { best = bd[m][i]; bi = bk[m][i]; }
    bk[m][0] = bi;
  }
  __syncthreads();
  float myl = 0.f;
  {
    int m = tid >> 4, dpart = tid & 15;
    int q = bk[m][0];
    int p = mtile * 16 + m;
    int b = p >> 12, rem = p & 4095;
    #pragma unroll
    for (int dd = 0; dd < 2; ++dd) {
      int d = dpart + dd * 16;
      float zq = csh[q * 32 + d];
      float df = zq - zsh[m][d];
      myl = fmaf(df, df, myl);
      enc[((b * 32 + d) << 12) + rem] = zq;          // z_q_st == z_q in value
    }
  }
  lsum[tid] = myl;
  __syncthreads();
  for (int s = 128; s > 0; s >>= 1) {
    if (tid < s) lsum[tid] += lsum[tid + s];
    __syncthreads();
  }
  if (tid == 0) atomicAdd(loss, lsum[0] * (1.0f / 2097152.0f)); // 2*sum/(131072*32)
}

// ---------------------------------------------------------------- deconv1 (WMMA f16)
// enc (32,32,64,64) -> d1 (32,16,128,128). ConvT k4 s2 p1, per output-parity class:
// K = 32ic*2ky*2kx = 128. Patch staged in LDS per wave; ds-only fragments.
__global__ void k_deconv1_wmma(const float* __restrict__ enc, const float* __restrict__ dw1,
                               const float* __restrict__ db1, float* __restrict__ out) {
  __shared__ _Float16 wsh[2][2][16][128];            // 16KB, per parity combo
  __shared__ _Float16 patch[8][32][2][18];           // 18KB: per-wave input patch
  for (int i = threadIdx.x; i < 2 * 2 * 16 * 128; i += blockDim.x) {
    int k = i & 127, oc = (i >> 7) & 15, pxv = (i >> 11) & 1, pyv = i >> 12;
    int ic = k >> 2, kyi = (k >> 1) & 1, kxi = k & 1;
    int ky = (pyv ? 0 : 1) + 2 * kyi;                // oy even -> ky in {1,3}
    int kx = (pxv ? 0 : 1) + 2 * kxi;
    wsh[pyv][pxv][oc][k] = (_Float16)dw1[((ic * 16 + oc) * 4 + ky) * 4 + kx];
  }
  const int lane = threadIdx.x & 31;
  const int wv   = threadIdx.x >> 5;
  const int tile = blockIdx.x * 8 + wv;              // 32768 tiles
  const int xg   = tile & 3;
  const int pxv  = (tile >> 2) & 1;
  const int oy   = (tile >> 3) & 127;
  const int b    = tile >> 10;
  const int pyv  = oy & 1;
  const int ky0  = pyv ? 0 : 1;
  const int kx0  = pxv ? 0 : 1;
  const int x0   = xg * 32 + pxv;
  const int iyTop = (oy + 1 - ky0) >> 1;             // r=kyi: iy = iyTop - r
  const int col0  = ((x0 + 1 - kx0) >> 1) - 1;       // local col = lm + 1 - kxi
  const float* ib = enc + (size_t)b * 32 * 64 * 64;
  // fill patch: 32ic x 2rows x 18cols (cols 0..16 used)
  for (int i = lane; i < 32 * 2 * 18; i += 32) {
    int col = i % 18; int t = i / 18; int r = t & 1; int ic = t >> 1;
    int iy = iyTop - r, ix = col0 + col;
    float v = 0.f;
    if ((unsigned)iy < 64u && (unsigned)ix < 64u)
      v = ib[(ic * 64 + iy) * 64 + ix];
    patch[wv][ic][r][col] = (_Float16)v;
  }
  __syncthreads();
  const int hi = lane >> 4, lm = lane & 15;
  const int oc = lm;
  v8f c = {};
  for (int kb = 0; kb < 128; kb += 32) {
    v16h a;
    #pragma unroll
    for (int j = 0; j < 16; ++j) {
      int k = kb + hi * 16 + j;
      int ic = k >> 2, kyi = (k >> 1) & 1, kxi = k & 1;
      a[j] = patch[wv][ic][kyi][lm + 1 - kxi];
    }
    v16h bf = *(const v16h*)&wsh[pyv][pxv][oc][kb + hi * 16];
    c = __builtin_amdgcn_wmma_f32_16x16x32_f16(false, a, false, bf, (short)0, c, false, false);
  }
  const float bias = db1[oc];
  #pragma unroll
  for (int j = 0; j < 8; ++j) {
    int m = j + hi * 8;
    int oxm = x0 + 2 * m;
    out[((b * 16 + oc) * 128 + oy) * 128 + oxm] = fmaxf(c[j] + bias, 0.0f);
  }
}

// ---------------------------------------------------------------- deconv2 (direct)
// d1 (32,16,128,128) -> decoded (32,3,256,256). K=64: direct VALU.
__global__ void k_deconv2(const float* __restrict__ d1, const float* __restrict__ dw2,
                          const float* __restrict__ db2, float* __restrict__ out) {
  __shared__ float ws[16 * 3 * 16];
  for (int i = threadIdx.x; i < 768; i += blockDim.x) ws[i] = dw2[i];
  __syncthreads();
  int idx = blockIdx.x * blockDim.x + threadIdx.x;   // 32*3*256*256
  int ox = idx & 255; int t = idx >> 8;
  int oy = t & 255;   t >>= 8;
  int oc = t % 3;     int b = t / 3;
  int ky0 = (oy + 1) & 1, kx0 = (ox + 1) & 1;
  float acc = db2[oc];
  const float* ib = d1 + (size_t)b * 16 * 128 * 128;
  #pragma unroll
  for (int kyi = 0; kyi < 2; ++kyi) {
    int ky = ky0 + 2 * kyi;
    int iy = (oy + 1 - ky) >> 1;
    if ((unsigned)iy >= 128u) continue;
    #pragma unroll
    for (int kxi = 0; kxi < 2; ++kxi) {
      int kx = kx0 + 2 * kxi;
      int ix = (ox + 1 - kx) >> 1;
      if ((unsigned)ix >= 128u) continue;
      #pragma unroll
      for (int ic = 0; ic < 16; ++ic)
        acc = fmaf(ib[(ic * 128 + iy) * 128 + ix], ws[((ic * 3 + oc) * 4 + ky) * 4 + kx], acc);
    }
  }
  out[idx] = fmaxf(acc, 0.0f);
}

__global__ void k_zero_loss(float* loss) { if (threadIdx.x == 0) *loss = 0.0f; }

extern "C" void kernel_launch(void* const* d_in, const int* in_sizes, int n_in,
                              void* d_out, int out_size, void* d_ws, size_t ws_size,
                              hipStream_t stream) {
  const float* imgs = (const float*)d_in[0];
  const float* w1   = (const float*)d_in[1];
  const float* b1   = (const float*)d_in[2];
  const float* w2   = (const float*)d_in[3];
  const float* b2   = (const float*)d_in[4];
  const float* cb   = (const float*)d_in[5];
  const float* dw1  = (const float*)d_in[6];
  const float* db1  = (const float*)d_in[7];
  const float* dw2  = (const float*)d_in[8];
  const float* db2  = (const float*)d_in[9];

  float* enc  = (float*)d_out;                       // 32*32*64*64   = 4194304
  float* dec  = (float*)d_out + 4194304;             // 32*3*256*256  = 6291456
  float* loss = (float*)d_out + 10485760;            // scalar

  float* h  = (float*)d_ws;                          // 8388608 f (32MB)
  float* ze = (float*)d_ws + 8388608;                // 4194304 f (16MB)
  float* d1 = (float*)d_ws;                          // reuse h after conv2

  k_zero_loss   <<<1,     32,  0, stream>>>(loss);
  k_conv1       <<<32768, 256, 0, stream>>>(imgs, w1, b1, h);
  k_conv2_wmma  <<<2048,  256, 0, stream>>>(h, w2, b2, ze);
  k_vq          <<<8192,  256, 0, stream>>>(ze, cb, enc, loss);
  k_deconv1_wmma<<<4096,  256, 0, stream>>>(enc, dw1, db1, d1);
  k_deconv2     <<<24576, 256, 0, stream>>>(d1, dw2, db2, dec);
}
